// SNN_Only_65214783422979
// MI455X (gfx1250) — matrix operand order repaired and checked by reading, not compiled, not run on hardware
//
#include <hip/hip_runtime.h>

typedef __attribute__((ext_vector_type(16))) _Float16 v16h;
typedef __attribute__((ext_vector_type(8)))  _Float16 v8h;
typedef __attribute__((ext_vector_type(8)))  float    v8f;

namespace {

constexpr int kIN = 128;
constexpr int kH1 = 256;
constexpr int kH2 = 128;
constexpr int kNC = 10;
constexpr float kBeta = 0.5f;
constexpr float kThr  = 1.0f;

__device__ __forceinline__ v8f wmma16(v16h a, v16h b, v8f c) {
  // D = A(16x32 f16) x B(32x16 f16) + C(16x16 f32)
  return __builtin_amdgcn_wmma_f32_16x16x32_f16(
      /*neg_a=*/false, a, /*neg_b=*/false, b,
      /*c_mod=*/(short)0, c, /*reuse_a=*/false, /*reuse_b=*/false);
}

// snntorch Leaky, reset_mechanism='subtract':
//   reset = (mem_prev > THR); mem = BETA*mem_prev + cur - reset*THR;
//   spk = (mem - THR) > 0
__device__ __forceinline__ v8f lif(v8f& mem, v8f cur) {
  v8f spk;
#pragma unroll
  for (int i = 0; i < 8; ++i) {
    float reset = (mem[i] > kThr) ? kThr : 0.0f;
    float m = kBeta * mem[i] + cur[i] - reset;
    spk[i] = (m > kThr) ? 1.0f : 0.0f;
    mem[i] = m;
  }
  return spk;
}

// 16 consecutive f32 -> one B-fragment lane slice (v16h). For B[k][n]=W[n][k],
// lane (n = l%16, hi = l>=16) needs W[n][kt*32 + hi*16 .. +15]: contiguous.
__device__ __forceinline__ v16h cvt16(const float* __restrict__ p) {
  v16h r;
#pragma unroll
  for (int i = 0; i < 16; ++i) r[i] = (_Float16)p[i];
  return r;
}

} // namespace

__global__ void __launch_bounds__(256, 1)
snn_fused_kernel(const float* __restrict__ x,
                 const float* __restrict__ W1, const float* __restrict__ b1,
                 const float* __restrict__ W2, const float* __restrict__ b2,
                 const float* __restrict__ Wo, const float* __restrict__ bo,
                 const int* __restrict__ nsteps,
                 float* __restrict__ out) {
  // Spike exchange buffers: row-major [16 rows][H] f16
  __shared__ _Float16 __attribute__((aligned(16))) s1sh[16 * kH1]; // 8 KB
  __shared__ _Float16 __attribute__((aligned(16))) s2sh[16 * kH2]; // 4 KB

  const int wave = threadIdx.x >> 5;   // 8 waves per WG
  const int lane = threadIdx.x & 31;
  const int ln   = lane & 15;
  const int hi   = lane >> 4;
  const int m0   = blockIdx.x * 16;    // this WG's 16 batch rows

  // ---- W2 B-fragments for this wave's H2 column tile (resident in VGPRs) ----
  const int n2 = wave * 16 + ln;       // H2 output column
  v16h w2f[8];
#pragma unroll
  for (int kt = 0; kt < 8; ++kt)
    w2f[kt] = cvt16(W2 + (size_t)n2 * kH1 + kt * 32 + hi * 16);
  const float bias2 = b2[n2];

  // ---- Wo B-fragments (layer 3 handled by wave 0 only; NC=10 padded to 16) ----
  v16h wof[4];
  float biaso = 0.0f;
  if (wave == 0) {
#pragma unroll
    for (int kt = 0; kt < 4; ++kt) {
      if (ln < kNC) {
        wof[kt] = cvt16(Wo + (size_t)ln * kH2 + kt * 32 + hi * 16);
      } else {
        v16h z;
#pragma unroll
        for (int i = 0; i < 16; ++i) z[i] = (_Float16)0.0f;
        wof[kt] = z;
      }
    }
    if (ln < kNC) biaso = bo[ln];
  }

  // ---- x A-fragments (16 rows x K=128 -> 4 k-tiles), loaded once ----
  // A layout: lane m (lo) holds K {0..15}, lane m (hi) holds K {16..31}
  v16h xaf[4];
  {
    const float* xr = x + (size_t)(m0 + ln) * kIN;
#pragma unroll
    for (int kt = 0; kt < 4; ++kt) {
      const float* p = xr + kt * 32 + hi * 16;
      v16h a;
#pragma unroll
      for (int i = 0; i < 8; ++i) {
        a[i]     = (_Float16)p[i];        // K run 1
        a[8 + i] = (_Float16)p[8 + i];    // K run 2 (+8)
      }
      xaf[kt] = a;
    }
  }

  // ---- Hoisted cur1 = x @ W1^T + b1 : this wave's two H1 n-tiles ----
  v8f cur1[2], m1[2];
#pragma unroll
  for (int t2 = 0; t2 < 2; ++t2) {
    const int n1 = (wave * 2 + t2) * 16 + ln;
    const float bb = b1[n1];
    v8f acc;
#pragma unroll
    for (int i = 0; i < 8; ++i) acc[i] = bb;
#pragma unroll
    for (int kt = 0; kt < 4; ++kt)
      acc = wmma16(xaf[kt], cvt16(W1 + (size_t)n1 * kIN + kt * 32 + hi * 16), acc);
    cur1[t2] = acc;
#pragma unroll
    for (int i = 0; i < 8; ++i) m1[t2][i] = 0.0f;
  }

  v8f m2, mo, cnt;
#pragma unroll
  for (int i = 0; i < 8; ++i) { m2[i] = 0.0f; mo[i] = 0.0f; cnt[i] = 0.0f; }

  const int T = nsteps[0];
  for (int t = 0; t < T; ++t) {
    // ---- LIF layer 1 (regs) -> s1 spikes into LDS, f16 row-major ----
#pragma unroll
    for (int t2 = 0; t2 < 2; ++t2) {
      v8f spk = lif(m1[t2], cur1[t2]);
      const int n1 = (wave * 2 + t2) * 16 + ln;
#pragma unroll
      for (int i = 0; i < 8; ++i)
        s1sh[(i + 8 * hi) * kH1 + n1] = (_Float16)spk[i];   // C-layout row i+8*hi
    }
    __syncthreads();

    // ---- Layer 2: cur2 = s1 @ W2^T + b2 (one 16x16 tile per wave) ----
    v8f acc2;
#pragma unroll
    for (int i = 0; i < 8; ++i) acc2[i] = bias2;
#pragma unroll
    for (int kt = 0; kt < 8; ++kt) {
      // A-fragment of s1 from LDS: two aligned 16B reads per lane
      const _Float16* q = s1sh + ln * kH1 + kt * 32 + hi * 16;
      v8h r1 = *(const v8h*)q;          // K run 1
      v8h r2 = *(const v8h*)(q + 8);    // K run 2
      v16h a;
#pragma unroll
      for (int i = 0; i < 8; ++i) { a[i] = r1[i]; a[8 + i] = r2[i]; }
      acc2 = wmma16(a, w2f[kt], acc2);
    }
    v8f spk2 = lif(m2, acc2);
#pragma unroll
    for (int i = 0; i < 8; ++i)
      s2sh[(i + 8 * hi) * kH2 + n2] = (_Float16)spk2[i];
    __syncthreads();

    // ---- Layer 3 (wave 0): curo = s2 @ Wo^T + bo; accumulate spike counts ----
    if (wave == 0) {
      v8f acc3;
#pragma unroll
      for (int i = 0; i < 8; ++i) acc3[i] = biaso;
#pragma unroll
      for (int kt = 0; kt < 4; ++kt) {
        const _Float16* q = s2sh + ln * kH2 + kt * 32 + hi * 16;
        v8h r1 = *(const v8h*)q;
        v8h r2 = *(const v8h*)(q + 8);
        v16h a;
#pragma unroll
        for (int i = 0; i < 8; ++i) { a[i] = r1[i]; a[8 + i] = r2[i]; }
        acc3 = wmma16(a, wof[kt], acc3);
      }
      cnt += lif(mo, acc3);
    }
  }

  // ---- Write spike counts [16 x 10] for this WG's batch rows ----
  if (wave == 0 && ln < kNC) {
#pragma unroll
    for (int i = 0; i < 8; ++i)
      out[(size_t)(m0 + i + 8 * hi) * kNC + ln] = cnt[i];
  }
}

extern "C" void kernel_launch(void* const* d_in, const int* in_sizes, int n_in,
                              void* d_out, int out_size, void* d_ws, size_t ws_size,
                              hipStream_t stream) {
  (void)n_in; (void)out_size; (void)d_ws; (void)ws_size;
  const float* x  = (const float*)d_in[0];
  const float* W1 = (const float*)d_in[1];
  const float* b1 = (const float*)d_in[2];
  const float* W2 = (const float*)d_in[3];
  const float* b2 = (const float*)d_in[4];
  const float* Wo = (const float*)d_in[5];
  const float* bo = (const float*)d_in[6];
  const int* nsteps = (const int*)d_in[7];
  float* out = (float*)d_out;

  const int B = in_sizes[0] / kIN;   // 16384 batch rows
  dim3 grid(B / 16);                  // one WG per 16 batch rows
  dim3 block(256);                    // 8 wave32
  hipLaunchKernelGGL(snn_fused_kernel, grid, block, 0, stream,
                     x, W1, b1, W2, b2, Wo, bo, nsteps, out);
}
